// GNN_58196806861203
// MI455X (gfx1250) — matrix operand order repaired and checked by reading, not compiled, not run on hardware
//
#include <hip/hip_runtime.h>
#include <hip/hip_bf16.h>

#define N_NODES 50000
#define N_EDGES 600000
#define DIM     128
#define DIM2    256
#define NLAYERS 5
#define BN_EPS  1e-5f

typedef __attribute__((ext_vector_type(16))) __bf16 v16bf;
typedef __attribute__((ext_vector_type(8)))  __bf16 v8bf;
typedef __attribute__((ext_vector_type(8)))  float  v8f;

// ---------------- embedding gather: h[i,:] = emb[x_nodes[i],:] ----------------
__global__ __launch_bounds__(256) void gin_embed(const int* __restrict__ xn,
                                                 const float* __restrict__ emb,
                                                 float* __restrict__ h) {
    int t = blockIdx.x * blockDim.x + threadIdx.x;   // N*32 threads
    int node = t >> 5;
    if (node >= N_NODES) return;
    int f = (t & 31) * 4;
    float4 v = *(const float4*)(emb + (size_t)xn[node] * DIM + f);
    *(float4*)(h + (size_t)node * DIM + f) = v;
}

// ---- pack W1/W2 (f32) into bf16 WMMA B-fragment layout, once per launch ------
// unit u = (((layer*CT + ct)*KB + kbi)*32 + lane) -> 16 bf16 (32B) for that lane.
// lane: n = ct*16 + (lane&15), K = kbi*32 + (lane>>4)*8 + j + (j>=8 ? 8 : 0)  (j = 0..15)
__global__ __launch_bounds__(256) void gin_pack_weights(const float* __restrict__ W1,
                                                        const float* __restrict__ W2,
                                                        v16bf* __restrict__ w1p,
                                                        v16bf* __restrict__ w2p) {
    int u = blockIdx.x * blockDim.x + threadIdx.x;   // 10240 units each
    if (u >= NLAYERS * 16 * 4 * 32) return;
    int lane = u & 31;
    int t    = u >> 5;
    int m  = lane & 15;
    int kh = lane >> 4;
    {   // W1: [128,256], 16 col tiles, 4 K-steps
        int kbi = t & 3;
        int ct  = (t >> 2) & 15;
        int l   = t >> 6;
        int n   = ct * 16 + m;
        int ks  = kbi * 32 + kh * 8;
        v16bf v;
#pragma unroll
        for (int j = 0; j < 16; ++j)
            v[j] = (__bf16)W1[((size_t)(l * DIM + ks + j + (j >= 8 ? 8 : 0))) * DIM2 + n];
        w1p[u] = v;
    }
    {   // W2: [256,128], 8 col tiles, 8 K-steps
        int kbi = t & 7;
        int ct  = (t >> 3) & 7;
        int l   = t >> 6;
        int n   = ct * 16 + m;
        int ks  = kbi * 32 + kh * 8;
        v16bf v;
#pragma unroll
        for (int j = 0; j < 16; ++j)
            v[j] = (__bf16)W2[((size_t)(l * DIM2 + ks + j + (j >= 8 ? 8 : 0))) * DIM + n];
        w2p[u] = v;
    }
}

// ---------------- agg = h (self loop term) ----------------
__global__ __launch_bounds__(256) void gin_copy(const float* __restrict__ h,
                                                float* __restrict__ agg) {
    int t = blockIdx.x * blockDim.x + threadIdx.x;   // N*32 threads
    int node = t >> 5;
    if (node >= N_NODES) return;
    int f = (t & 31) * 4;
    *(float4*)(agg + (size_t)node * DIM + f) = *(const float4*)(h + (size_t)node * DIM + f);
}

// ---------------- edge scatter: agg[dst] += h[src] (L2-resident f32 atomics) --
__global__ __launch_bounds__(256) void gin_scatter(const int* __restrict__ src,
                                                   const int* __restrict__ dst,
                                                   const float* __restrict__ h,
                                                   float* __restrict__ agg) {
    int t = blockIdx.x * blockDim.x + threadIdx.x;   // E*32 threads
    int e = t >> 5;
    if (e >= N_EDGES) return;
    int f = (t & 31) * 4;
    int s = src[e], d = dst[e];
    float4 v = *(const float4*)(h + (size_t)s * DIM + f);
    float* p = agg + (size_t)d * DIM + f;
    atomicAdd(p + 0, v.x);
    atomicAdd(p + 1, v.y);
    atomicAdd(p + 2, v.z);
    atomicAdd(p + 3, v.w);
}

// -------- GEMM1: z1 = relu(agg[N,128] @ W1[128,256] + b1) -> A-fragment layout -
// Block = 4 waves. Wave w computes rows [16*bx,16*bx+16) x cols [64w, 64w+64):
// one A fragment feeds 4 WMMAs per K-step. Output staged through per-wave LDS
// and emitted in GEMM2 A-fragment layout (32B contiguous per lane).
__global__ __launch_bounds__(128) void gin_gemm1(const float* __restrict__ agg,
                                                 const v16bf* __restrict__ w1p,
                                                 const float* __restrict__ b1,
                                                 v16bf* __restrict__ z1p,
                                                 int layer) {
    __shared__ __attribute__((aligned(16))) __bf16 stage[4][16][72];  // 72 = 64 + pad
    const int lane  = threadIdx.x & 31;
    const int wave  = threadIdx.x >> 5;
    const int row0  = blockIdx.x * 16;              // 3125 blocks exactly cover N
    const int m     = lane & 15;
    const int khalf = lane >> 4;

    v8f acc0 = {}, acc1 = {}, acc2 = {}, acc3 = {};
    for (int kbi = 0; kbi < 4; ++kbi) {
        const int kb = kbi * 32;
        // A fragment (16x32 bf16 from f32): lane = row m, K split at 8 by lane-half
        const float* arow = agg + (size_t)(row0 + m) * DIM + kb + khalf * 8;
        float4 a0 = *(const float4*)(arow + 0);
        float4 a1 = *(const float4*)(arow + 4);
        float4 a2 = *(const float4*)(arow + 16);
        float4 a3 = *(const float4*)(arow + 20);
        v16bf A;
        A[0]  = (__bf16)a0.x; A[1]  = (__bf16)a0.y; A[2]  = (__bf16)a0.z; A[3]  = (__bf16)a0.w;
        A[4]  = (__bf16)a1.x; A[5]  = (__bf16)a1.y; A[6]  = (__bf16)a1.z; A[7]  = (__bf16)a1.w;
        A[8]  = (__bf16)a2.x; A[9]  = (__bf16)a2.y; A[10] = (__bf16)a2.z; A[11] = (__bf16)a2.w;
        A[12] = (__bf16)a3.x; A[13] = (__bf16)a3.y; A[14] = (__bf16)a3.z; A[15] = (__bf16)a3.w;

        const size_t wb = ((size_t)(layer * 16 + wave * 4) * 4 + kbi) * 32 + lane;
        v16bf B0 = w1p[wb + 0 * 4 * 32];            // ct = 4w+0 .. 4w+3
        v16bf B1 = w1p[wb + 1 * 4 * 32];
        v16bf B2 = w1p[wb + 2 * 4 * 32];
        v16bf B3 = w1p[wb + 3 * 4 * 32];
        acc0 = __builtin_amdgcn_wmma_f32_16x16x32_bf16(false, A, false, B0, (short)0, acc0, false, false);
        acc1 = __builtin_amdgcn_wmma_f32_16x16x32_bf16(false, A, false, B1, (short)0, acc1, false, false);
        acc2 = __builtin_amdgcn_wmma_f32_16x16x32_bf16(false, A, false, B2, (short)0, acc2, false, false);
        acc3 = __builtin_amdgcn_wmma_f32_16x16x32_bf16(false, A, false, B3, (short)0, acc3, false, false);
    }

    // bias + ReLU, stage 16x64 bf16 tile in this wave's LDS region (row-major)
    const v8f accs[4] = {acc0, acc1, acc2, acc3};
#pragma unroll
    for (int c = 0; c < 4; ++c) {
        const float bias = b1[layer * DIM2 + wave * 64 + c * 16 + m];
#pragma unroll
        for (int r = 0; r < 8; ++r) {
            float v = accs[c][r] + bias;
            v = v > 0.f ? v : 0.f;
            stage[wave][r + khalf * 8][c * 16 + m] = (__bf16)v;   // ds_store_b16
        }
    }

    // re-read in GEMM2 A-fragment order (same wave: no barrier, dscnt only) and
    // emit z1 in fragment layout: unit = (rowTile*8 + kbi)*32 + lane, 32B/lane.
#pragma unroll
    for (int kbl = 0; kbl < 2; ++kbl) {             // this wave owns kbi = 2w+kbl
        const __bf16* p = &stage[wave][m][kbl * 32 + khalf * 8];
        v8bf g0 = *(const v8bf*)(p);                // ds_load_b128
        v8bf g1 = *(const v8bf*)(p + 16);
        v16bf out;
#pragma unroll
        for (int j = 0; j < 8; ++j) { out[j] = g0[j]; out[8 + j] = g1[j]; }
        z1p[((size_t)blockIdx.x * 8 + wave * 2 + kbl) * 32 + lane] = out;
    }
}

// -------- GEMM2: z2 = z1[N,256] @ W2[256,128] + b2, + fused BN stats ----------
// Block = 4 waves; wave w covers cols [32w, 32w+32) (2 tiles). A fragment is a
// single coalesced 32B/lane load from the fragment-layout z1.
__global__ __launch_bounds__(128) void gin_gemm2(const v16bf* __restrict__ z1p,
                                                 const v16bf* __restrict__ w2p,
                                                 const float* __restrict__ b2,
                                                 float* __restrict__ z2,
                                                 float* __restrict__ colsum,
                                                 float* __restrict__ colsq,
                                                 int layer) {
    const int lane  = threadIdx.x & 31;
    const int wave  = threadIdx.x >> 5;
    const int row0  = blockIdx.x * 16;
    const int m     = lane & 15;
    const int khalf = lane >> 4;

    v8f acc0 = {}, acc1 = {};
    for (int kbi = 0; kbi < 8; ++kbi) {
        v16bf A = z1p[((size_t)blockIdx.x * 8 + kbi) * 32 + lane];
        const size_t wb = ((size_t)(layer * 8 + wave * 2) * 8 + kbi) * 32 + lane;
        v16bf B0 = w2p[wb + 0 * 8 * 32];            // ct = 2w+0, 2w+1
        v16bf B1 = w2p[wb + 1 * 8 * 32];
        acc0 = __builtin_amdgcn_wmma_f32_16x16x32_bf16(false, A, false, B0, (short)0, acc0, false, false);
        acc1 = __builtin_amdgcn_wmma_f32_16x16x32_bf16(false, A, false, B1, (short)0, acc1, false, false);
    }

    const v8f accs[2] = {acc0, acc1};
#pragma unroll
    for (int c = 0; c < 2; ++c) {
        const int n = wave * 32 + c * 16 + m;
        const float bias = b2[layer * DIM + n];
        float s = 0.f, ss = 0.f;
#pragma unroll
        for (int r = 0; r < 8; ++r) {
            float v = accs[c][r] + bias;
            z2[(size_t)(row0 + r + khalf * 8) * DIM + n] = v;
            s  += v;
            ss += v * v;
        }
        atomicAdd(&colsum[n], s);                   // lane's 8 values share column n
        atomicAdd(&colsq[n], ss);
    }
}

// ---------------- zero BN stat accumulators (colsum|colsq contiguous) ----------
__global__ void gin_zero_stats(float* __restrict__ s) { s[threadIdx.x] = 0.f; }

// ---------------- BN (training-mode batch stats) + optional ReLU -> h ----------
__global__ __launch_bounds__(256) void gin_bn(const float* __restrict__ z2,
                                              const float* __restrict__ colsum,
                                              const float* __restrict__ colsq,
                                              const float* __restrict__ gamma,
                                              const float* __restrict__ beta,
                                              float* __restrict__ h,
                                              int layer, int do_relu) {
    int t = blockIdx.x * blockDim.x + threadIdx.x;   // N*32 threads
    int node = t >> 5;
    if (node >= N_NODES) return;
    int f = (t & 31) * 4;
    const float invN = 1.0f / (float)N_NODES;
    float4 z = *(const float4*)(z2 + (size_t)node * DIM + f);
    float out[4] = {z.x, z.y, z.z, z.w};
#pragma unroll
    for (int c = 0; c < 4; ++c) {
        int n = f + c;
        float mean = colsum[n] * invN;
        float var  = colsq[n] * invN - mean * mean;
        float y = gamma[layer * DIM + n] * (out[c] - mean) * rsqrtf(var + BN_EPS)
                  + beta[layer * DIM + n];
        if (do_relu) y = y > 0.f ? y : 0.f;
        out[c] = y;
    }
    float4 o = {out[0], out[1], out[2], out[3]};
    *(float4*)(h + (size_t)node * DIM + f) = o;
}

extern "C" void kernel_launch(void* const* d_in, const int* in_sizes, int n_in,
                              void* d_out, int out_size, void* d_ws, size_t ws_size,
                              hipStream_t stream) {
    (void)in_sizes; (void)n_in; (void)out_size; (void)ws_size;
    const int*   xn    = (const int*)d_in[0];
    const int*   src   = (const int*)d_in[1];
    const int*   dst   = (const int*)d_in[2];
    const float* emb   = (const float*)d_in[3];
    const float* W1    = (const float*)d_in[4];
    const float* b1    = (const float*)d_in[5];
    const float* W2    = (const float*)d_in[6];
    const float* b2    = (const float*)d_in[7];
    const float* gamma = (const float*)d_in[8];
    const float* beta  = (const float*)d_in[9];
    float* h = (float*)d_out;                       // h lives in d_out across layers

    // workspace: agg f32 | z1 (fragment layout) | z2 f32 | w1p | w2p | stats
    char*   ws     = (char*)d_ws;
    float*  agg    = (float*)(ws);                          // 25,600,000 B
    v16bf*  z1p    = (v16bf*)(ws + 25600000);               // 3125*8*32*32B = 25,600,000 B
    float*  z2     = (float*)(ws + 51200000);               // 25,600,000 B
    v16bf*  w1p    = (v16bf*)(ws + 76800000);               // 10240*32B = 327,680 B
    v16bf*  w2p    = (v16bf*)(ws + 77127680);               // 327,680 B
    float*  colsum = (float*)(ws + 77455360);               // 128 f32
    float*  colsq  = colsum + DIM;                          // 128 f32 (contiguous)

    const int nodeBlocks = (N_NODES * 32) / 256;            // 6250 (exact)
    const int edgeBlocks = (N_EDGES * 32) / 256;            // 75000 (exact)

    gin_embed<<<nodeBlocks, 256, 0, stream>>>(xn, emb, h);
    gin_pack_weights<<<40, 256, 0, stream>>>(W1, W2, w1p, w2p);   // 10240 units

    for (int l = 0; l < NLAYERS; ++l) {
        gin_copy<<<nodeBlocks, 256, 0, stream>>>(h, agg);
        gin_scatter<<<edgeBlocks, 256, 0, stream>>>(src, dst, h, agg);
        gin_gemm1<<<N_NODES / 16, 128, 0, stream>>>(agg, w1p, b1, z1p, l);
        gin_zero_stats<<<1, 256, 0, stream>>>(colsum);
        gin_gemm2<<<N_NODES / 16, 128, 0, stream>>>(z1p, w2p, b2, z2, colsum, colsq, l);
        gin_bn<<<nodeBlocks, 256, 0, stream>>>(z2, colsum, colsq, gamma, beta, h, l,
                                               (l < NLAYERS - 1) ? 1 : 0);
    }
}
